// QuantizedLinear4bit_53626961657867
// MI455X (gfx1250) — compile-verified, matrix operand order checked by reading, not run on hardware
//
#include <hip/hip_runtime.h>

typedef __attribute__((ext_vector_type(16))) __bf16           v16bf;
typedef __attribute__((ext_vector_type(8)))  float            v8f;
typedef __attribute__((ext_vector_type(8)))  unsigned short   u16x8;

#define MTOT 4096   // B*S = 2*2048
#define NTOT 4096   // OF
#define KTOT 4096   // IF
#define BM   256
#define BN   128
#define BK   32
#define LDA  40     // ushort row stride in LDS: 32 + 8 pad -> 80B (16B aligned, conflict-free)

// packed f32x2 -> bf16x2 (RNE) in one VALU op
__device__ __forceinline__ unsigned int cvt_pk_bf16(float a, float b) {
    unsigned int r;
    asm volatile("v_cvt_pk_bf16_f32 %0, %1, %2" : "=v"(r) : "v"(a), "v"(b));
    return r;   // low 16 = cvt(a), high 16 = cvt(b)
}

__global__ __launch_bounds__(256, 1)
void QuantizedLinear4bit_wmma_v3(const float* __restrict__ x,
                                 const int*   __restrict__ Wp,
                                 const float* __restrict__ SZ,
                                 const float* __restrict__ bias,
                                 unsigned short* __restrict__ out)
{
    __shared__ __align__(16) unsigned short As[BM * LDA];  // 20480 B: bf16 x tile [m][k]
    __shared__ __align__(16) unsigned short Bs[BN * LDA];  // 10240 B: bf16 dequant W tile [n][k]
    __shared__ float Ss[BN];    // scale
    __shared__ float ZSs[BN];   // -zero*scale

    const int tid  = threadIdx.x;
    const int lane = tid & 31;
    const int wave = tid >> 5;       // 8 waves
    const int wm   = wave & 3;       // 4 M-groups of 64 rows
    const int wn   = wave >> 2;      // 2 N-groups of 64 cols
    const int half = lane >> 4;      // K-half selector (ISA fragment layout)
    const int l16  = lane & 15;

    const int nBase = blockIdx.x * BN;
    const int mBase = blockIdx.y * BM;

    // x staging: 8 lanes cover one row's 32 floats (coalesced); 8 passes -> 256 rows
    const int xr = tid >> 3;             // 0..31 base row
    const int xc = (tid & 7) * 4;        // float index within k-tile
    const float* xThread = x + (size_t)(mBase + xr) * KTOT + xc;

    // W staging: 32 packed rows x 16 k-pairs = 512 pairs; 2 per thread
    const int kp2   = (tid & 15) * 2;    // even k within tile
    const int rl0   = tid >> 4;          // 0..15 packed row
    const int rBase = nBase >> 2;
    const int* wThread = Wp + (size_t)(rBase + rl0) * KTOT + kp2;

    v8f acc[4][4];
    #pragma unroll
    for (int i = 0; i < 4; ++i)
        #pragma unroll
        for (int j = 0; j < 4; ++j) { v8f z = {}; acc[i][j] = z; }

    // ---- software pipeline registers: globals for the tile about to be staged ----
    float4 xf[8];
    int2   wv[2];

    // prologue: load tile kt=0 into registers
    #pragma unroll
    for (int p = 0; p < 8; ++p)
        xf[p] = *(const float4*)(xThread + (size_t)(32 * p) * KTOT);
    #pragma unroll
    for (int p = 0; p < 2; ++p)
        wv[p] = *(const int2*)(wThread + (size_t)(16 * p) * KTOT);

    for (int kt = 0; kt < KTOT; kt += BK) {
        // per-group scale / (-zero*scale), refreshed every group_size=128 (4 K-tiles).
        // Safe slot: previous compute phase never reads Ss/ZSs; staging reads after barrier.
        if (((kt & 127) == 0) && tid < 128) {
            const float* sz = SZ + (size_t)(kt >> 7) * (2 * NTOT) + 2 * (nBase + tid);
            float2 v = *(const float2*)sz;           // {scale, zero}
            Ss[tid]  = v.x;
            ZSs[tid] = -v.y * v.x;
        }
        __syncthreads();   // previous compute done reading LDS; Ss/ZSs visible

        // ---- stage x from registers: fp32 -> bf16 (v_cvt_pk_bf16_f32) -> As ----
        #pragma unroll
        for (int p = 0; p < 8; ++p) {
            unsigned int lo = cvt_pk_bf16(xf[p].x, xf[p].y);
            unsigned int hi = cvt_pk_bf16(xf[p].z, xf[p].w);
            *(uint2*)(&As[(32 * p + xr) * LDA + xc]) = make_uint2(lo, hi);
        }

        // ---- stage W from registers: nibble dequant via fma, packed bf16 store ----
        #pragma unroll
        for (int p = 0; p < 2; ++p) {
            int rl = rl0 + 16 * p;                                   // 0..31
            int2 v = wv[p];                                          // words for k, k+1
            #pragma unroll
            for (int j = 0; j < 4; ++j) {
                float q0 = (float)((v.x >> (4 * j)) & 15);
                float q1 = (float)((v.y >> (4 * j)) & 15);
                int   nl = 4 * rl + j;
                float s  = Ss[nl], zs = ZSs[nl];
                unsigned int pk = cvt_pk_bf16(fmaf(q0, s, zs), fmaf(q1, s, zs));
                *(unsigned int*)(&Bs[nl * LDA + kp2]) = pk;
            }
        }
        __syncthreads();

        // ---- issue global loads for NEXT tile (latency hidden under WMMA) ----
        if (kt + BK < KTOT) {
            const float* xp = xThread + kt + BK;
            #pragma unroll
            for (int p = 0; p < 8; ++p)
                xf[p] = *(const float4*)(xp + (size_t)(32 * p) * KTOT);
            const int* wp = wThread + kt + BK;
            #pragma unroll
            for (int p = 0; p < 2; ++p)
                wv[p] = *(const int2*)(wp + (size_t)(16 * p) * KTOT);
            __builtin_prefetch(xp + BK, 0, 1);   // global_prefetch_b8 one tile further
        }

        // ---- fragments + 16x v_wmma_f32_16x16x32_bf16 ----
        v16bf a[4];
        #pragma unroll
        for (int fm = 0; fm < 4; ++fm) {
            const unsigned short* p = &As[(wm * 64 + fm * 16 + l16) * LDA + half * 8];
            union { u16x8 h[2]; v16bf v; } u;
            u.h[0] = *(const u16x8*)p;          // K = 8*half .. +7
            u.h[1] = *(const u16x8*)(p + 16);   // K = 16 + 8*half .. +7
            a[fm] = u.v;
        }
        #pragma unroll
        for (int fn = 0; fn < 4; ++fn) {
            const unsigned short* p = &Bs[(wn * 64 + fn * 16 + l16) * LDA + half * 8];
            union { u16x8 h[2]; v16bf v; } u;
            u.h[0] = *(const u16x8*)p;
            u.h[1] = *(const u16x8*)(p + 16);
            v16bf b = u.v;
            #pragma unroll
            for (int fm = 0; fm < 4; ++fm)
                acc[fm][fn] = __builtin_amdgcn_wmma_f32_16x16x32_bf16(
                    false, a[fm], false, b, (short)0, acc[fm][fn], false, false);
        }
    }

    // ---- epilogue: bias add, packed bf16 convert, b16 stores ----
    #pragma unroll
    for (int fm = 0; fm < 4; ++fm) {
        #pragma unroll
        for (int fn = 0; fn < 4; ++fn) {
            int   col = nBase + wn * 64 + fn * 16 + l16;
            float bv  = bias[col];
            #pragma unroll
            for (int v = 0; v < 8; v += 2) {
                float f0 = acc[fm][fn][v]     + bv;
                float f1 = acc[fm][fn][v + 1] + bv;
                unsigned int pk = cvt_pk_bf16(f0, f1);
                size_t row = (size_t)(mBase + wm * 64 + fm * 16 + half * 8 + v);
                out[row * NTOT + col]       = (unsigned short)(pk & 0xFFFFu);
                out[(row + 1) * NTOT + col] = (unsigned short)(pk >> 16);
            }
        }
    }
}

extern "C" void kernel_launch(void* const* d_in, const int* in_sizes, int n_in,
                              void* d_out, int out_size, void* d_ws, size_t ws_size,
                              hipStream_t stream) {
    (void)in_sizes; (void)n_in; (void)out_size; (void)d_ws; (void)ws_size;
    const float* x    = (const float*)d_in[0];
    const int*   Wp   = (const int*)d_in[1];
    const float* SZ   = (const float*)d_in[2];
    const float* bias = (const float*)d_in[3];
    unsigned short* out = (unsigned short*)d_out;   // bf16 bit pattern

    dim3 grid(NTOT / BN, MTOT / BM);   // 32 x 16 workgroups
    dim3 block(256);
    QuantizedLinear4bit_wmma_v3<<<grid, block, 0, stream>>>(x, Wp, SZ, bias, out);
}
